// GaborAutoencoder_65163243815162
// MI455X (gfx1250) — matrix operand order repaired and checked by reading, not compiled, not run on hardware
//
#include <hip/hip_runtime.h>

// ---------------------------------------------------------------------------
// Types for gfx1250 WMMA (wave32): v_wmma_f32_16x16x32_bf16
// ---------------------------------------------------------------------------
typedef __bf16 bf16_t;
typedef __bf16 v4bf  __attribute__((ext_vector_type(4)));
typedef __bf16 v8bf  __attribute__((ext_vector_type(8)));
typedef __bf16 v16bf __attribute__((ext_vector_type(16)));
typedef float  v8f   __attribute__((ext_vector_type(8)));
typedef int    v4i   __attribute__((ext_vector_type(4)));

#define AS1 __attribute__((address_space(1)))
#define AS3 __attribute__((address_space(3)))

// Combine two contiguous 8 x bf16 (16B) loads into one WMMA fragment.
__device__ __forceinline__ v16bf frag_from(const bf16_t* __restrict__ p0,
                                           const bf16_t* __restrict__ p1) {
    v8bf lo = *(const v8bf*)p0;
    v8bf hi = *(const v8bf*)p1;
    return __builtin_shufflevector(lo, hi, 0,1,2,3,4,5,6,7,8,9,10,11,12,13,14,15);
}

// 16-byte global -> LDS copy. gfx1250 async-to-LDS path (tracked by ASYNCcnt);
// builtin signature (from probe): (v4i __device__*, v4i __shared__*, imm, imm).
__device__ __forceinline__ void copy16_g2l(const bf16_t* g, bf16_t* l) {
#if __has_builtin(__builtin_amdgcn_global_load_async_to_lds_b128)
    __builtin_amdgcn_global_load_async_to_lds_b128(
        (AS1 v4i*)const_cast<bf16_t*>(g), (AS3 v4i*)l, /*offset=*/0, /*cpol=*/0);
#else
    *(v8bf*)l = *(const v8bf*)g;
#endif
}

__device__ __forceinline__ void wait_g2l() {
#if __has_builtin(__builtin_amdgcn_global_load_async_to_lds_b128) && \
    __has_builtin(__builtin_amdgcn_s_wait_asynccnt)
    __builtin_amdgcn_s_wait_asynccnt(0);
#endif
}

// ---------------------------------------------------------------------------
// fp32 -> bf16 conversion (vectorized, n must be a multiple of 4)
// ---------------------------------------------------------------------------
__global__ __launch_bounds__(256)
void cvt_f32_bf16(const float* __restrict__ in, bf16_t* __restrict__ out, int n4) {
    int i = blockIdx.x * blockDim.x + threadIdx.x;
    if (i < n4) {
        float4 v = ((const float4*)in)[i];
        v4bf r;
        r[0] = (bf16_t)v.x; r[1] = (bf16_t)v.y; r[2] = (bf16_t)v.z; r[3] = (bf16_t)v.w;
        ((v4bf*)out)[i] = r;
    }
}

// ---------------------------------------------------------------------------
// LDS-tiled WMMA GEMM:  C = act(A @ W^T + bias), bf16 in / bf16 out.
//   A: (M,K) row-major, W: (N,K) row-major, bias: (N) f32.
// Block: 256 threads = 8 waves in a 2(M) x 4(N) grid; block tile 64 x 256;
// each wave owns a 32x64 tile (2x4 accumulators). K staged in 32-wide slabs,
// double-buffered in LDS via async global->LDS copies.
//
// Fragment layouts (wave32, 16-bit):
//   A (16x32 MxK): lane l holds row m=l%16, K = h*8+{0..7} and 16+h*8+{0..7}
//   B (32x16 KxN): lane l holds col n=l%16, K = h*8+{0..7} and 16+h*8+{0..7} (h = l>>4)
//   C (16x16 f32): vgpr r -> M = r + 8*h, N = l%16
// ---------------------------------------------------------------------------
#define MBLK 64
#define NBLK 256
#define KBLK 32
#define LDSS 48   // LDS row stride in elements (96B): 16B-aligned, low-conflict

template<bool RELU>
__global__ __launch_bounds__(256)
void gemm_bf16_wmma_lds(const bf16_t* __restrict__ A,
                        const bf16_t* __restrict__ W,
                        const float*  __restrict__ bias,
                        bf16_t* __restrict__ out,
                        int M, int N, int K) {
    __shared__ __align__(16) bf16_t sA[2][MBLK * LDSS];   //  2 x  6 KiB
    __shared__ __align__(16) bf16_t sB[2][NBLK * LDSS];   //  2 x 24 KiB

    const int tid  = threadIdx.x;
    const int lane = tid & 31;
    const int wave = tid >> 5;
    const int l16  = lane & 15;
    const int hl   = lane >> 4;
    const int wm   = wave >> 2;     // 0..1
    const int wn   = wave & 3;      // 0..3

    const int tilesN = N / NBLK;
    const int m_blk  = (blockIdx.x / tilesN) * MBLK;
    const int n_blk  = (blockIdx.x % tilesN) * NBLK;

    // Copy-task mapping: 256 threads -> 64 rows x 4 chunks of 16B.
    const int crow = tid >> 2;          // 0..63
    const int cch  = (tid & 3) * 8;     // element offset of this 16B chunk

    const bf16_t* gA = A + (size_t)(m_blk + crow) * (size_t)K + cch;
    const bf16_t* gB = W + (size_t)(n_blk + crow) * (size_t)K + cch;

    v8f acc[2][4];
#pragma unroll
    for (int i = 0; i < 2; ++i)
#pragma unroll
        for (int j = 0; j < 4; ++j)
#pragma unroll
            for (int r = 0; r < 8; ++r) acc[i][j][r] = 0.0f;

    auto load_tiles = [&](int buf, int k) {
        copy16_g2l(gA + k, &sA[buf][crow * LDSS + cch]);
#pragma unroll
        for (int i = 0; i < 4; ++i)
            copy16_g2l(gB + (size_t)(64 * i) * (size_t)K + k,
                       &sB[buf][(crow + 64 * i) * LDSS + cch]);
    };

    load_tiles(0, 0);
    wait_g2l();
    __syncthreads();

    int cur = 0;
    for (int k = 0; k < K; k += KBLK) {
        const bool more = (k + KBLK) < K;
        if (more) load_tiles(cur ^ 1, k + KBLK);

        v16bf af[2], bfr[4];
#pragma unroll
        for (int i = 0; i < 2; ++i) {
            const bf16_t* base = &sA[cur][(wm * 32 + i * 16 + l16) * LDSS];
            af[i] = frag_from(base + hl * 8, base + 16 + hl * 8);
        }
#pragma unroll
        for (int j = 0; j < 4; ++j) {
            const bf16_t* base = &sB[cur][(wn * 64 + j * 16 + l16) * LDSS];
            bfr[j] = frag_from(base + hl * 8, base + 16 + hl * 8);
        }
#pragma unroll
        for (int i = 0; i < 2; ++i)
#pragma unroll
            for (int j = 0; j < 4; ++j)
                acc[i][j] = __builtin_amdgcn_wmma_f32_16x16x32_bf16(
                    false, af[i], false, bfr[j],
                    (short)0, acc[i][j], false, false);

        if (more) wait_g2l();
        __syncthreads();
        if (more) cur ^= 1;
    }

    // Epilogue: bias + ReLU, bf16 stores per C layout.
    const int n0 = n_blk + wn * 64;
    const int m0 = m_blk + wm * 32;
#pragma unroll
    for (int j = 0; j < 4; ++j) {
        const int col = n0 + j * 16 + l16;
        const float bv = bias[col];
#pragma unroll
        for (int i = 0; i < 2; ++i) {
            const int mbase = m0 + i * 16 + hl * 8;
#pragma unroll
            for (int r = 0; r < 8; ++r) {
                float v = acc[i][j][r] + bv;
                if (RELU) v = fmaxf(v, 0.0f);
                out[(size_t)(mbase + r) * (size_t)N + (size_t)col] = (bf16_t)v;
            }
        }
    }
}

// ---------------------------------------------------------------------------
// Direct (register-only) WMMA GEMM for the small final layer (N=160).
// One wave per block, (16*MT) x (16*NT) tile.
// ---------------------------------------------------------------------------
template<int MT, int NT, bool RELU, bool OUT_BF16>
__global__ __launch_bounds__(32)
void gemm_bf16_wmma(const bf16_t* __restrict__ A,
                    const bf16_t* __restrict__ W,
                    const float*  __restrict__ bias,
                    void* __restrict__ out,
                    int M, int N, int K) {
    const int lane = threadIdx.x;
    const int l16  = lane & 15;
    const int hl   = lane >> 4;

    const int tilesN = N / (16 * NT);
    const int tm = blockIdx.x / tilesN;
    const int tn = blockIdx.x % tilesN;
    const int m0 = tm * 16 * MT;
    const int n0 = tn * 16 * NT;

    v8f acc[MT][NT];
#pragma unroll
    for (int i = 0; i < MT; ++i)
#pragma unroll
        for (int j = 0; j < NT; ++j)
#pragma unroll
            for (int r = 0; r < 8; ++r) acc[i][j][r] = 0.0f;

    const bf16_t* arow[MT];
#pragma unroll
    for (int i = 0; i < MT; ++i)
        arow[i] = A + (size_t)(m0 + i * 16 + l16) * (size_t)K;
    const bf16_t* wrow[NT];
#pragma unroll
    for (int j = 0; j < NT; ++j)
        wrow[j] = W + (size_t)(n0 + j * 16 + l16) * (size_t)K;

    for (int k = 0; k < K; k += 32) {
        v16bf afrag[MT], bfrag[NT];
#pragma unroll
        for (int i = 0; i < MT; ++i)
            afrag[i] = frag_from(arow[i] + k + hl * 8, arow[i] + k + 16 + hl * 8);
#pragma unroll
        for (int j = 0; j < NT; ++j)
            bfrag[j] = frag_from(wrow[j] + k + hl * 8, wrow[j] + k + 16 + hl * 8);
#pragma unroll
        for (int i = 0; i < MT; ++i)
#pragma unroll
            for (int j = 0; j < NT; ++j)
                acc[i][j] = __builtin_amdgcn_wmma_f32_16x16x32_bf16(
                    false, afrag[i], false, bfrag[j],
                    (short)0, acc[i][j], false, false);
    }

#pragma unroll
    for (int j = 0; j < NT; ++j) {
        const int col = n0 + j * 16 + l16;
        const float bv = bias[col];
#pragma unroll
        for (int i = 0; i < MT; ++i) {
            const int mbase = m0 + i * 16 + hl * 8;
#pragma unroll
            for (int r = 0; r < 8; ++r) {
                float v = acc[i][j][r] + bv;
                if (RELU) v = fmaxf(v, 0.0f);
                const size_t idx = (size_t)(mbase + r) * (size_t)N + (size_t)col;
                if (OUT_BF16) ((bf16_t*)out)[idx] = (bf16_t)v;
                else          ((float*)out)[idx]  = v;
            }
        }
    }
}

// ---------------------------------------------------------------------------
// Gabor synthesis: one block per batch row; 32 wavelet params -> LDS, each of
// 256 threads evaluates 8 time samples. v_exp_f32 / v_cos_f32 bound.
// ---------------------------------------------------------------------------
#define SIG_LEN   2048
#define N_WVL     32
#define TWO_PI_F  6.28318530717958647692f

__global__ __launch_bounds__(256)
void gabor_synth(const float* __restrict__ params, float* __restrict__ out) {
    __shared__ float sAm[N_WVL], sT0[N_WVL], sWv[N_WVL], sNH[N_WVL], sPh[N_WVL];
    const int b   = blockIdx.x;
    const int tid = threadIdx.x;

    if (tid < N_WVL) {
        const float* p = params + (size_t)b * (N_WVL * 5) + tid * 5;
        const float t0 = __fdividef((float)SIG_LEN, 1.0f + __expf(-p[1]));
        const float f  = __fdividef(0.5f,           1.0f + __expf(-p[2]));
        const float sg = __fdividef(200.0f,         1.0f + __expf(-p[3])) + 2.0f;
        sAm[tid] = p[0];
        sT0[tid] = t0;
        sWv[tid] = TWO_PI_F * f;
        sNH[tid] = -__fdividef(0.5f, sg * sg);
        sPh[tid] = p[4];
    }
    __syncthreads();

    float* __restrict__ o0 = out + (size_t)b * (2 * SIG_LEN);
    for (int it = 0; it < SIG_LEN / 256; ++it) {
        const int   t  = tid + it * 256;
        const float tf = (float)t;
        float sum = 0.0f;
#pragma unroll
        for (int n = 0; n < N_WVL; ++n) {
            const float dt = tf - sT0[n];
            sum += sAm[n] * __expf(sNH[n] * dt * dt)
                          * __cosf(__fmaf_rn(sWv[n], dt, sPh[n]));
        }
        o0[t]           = sum;
        o0[SIG_LEN + t] = sum;
    }
}

// ---------------------------------------------------------------------------
// Launch
// ---------------------------------------------------------------------------
extern "C" void kernel_launch(void* const* d_in, const int* in_sizes, int n_in,
                              void* d_out, int out_size, void* d_ws, size_t ws_size,
                              hipStream_t stream) {
    (void)in_sizes; (void)n_in; (void)out_size; (void)ws_size;

    const float* x  = (const float*)d_in[0];
    const float* W1 = (const float*)d_in[1];
    const float* b1 = (const float*)d_in[2];
    const float* W2 = (const float*)d_in[3];
    const float* b2 = (const float*)d_in[4];
    const float* W3 = (const float*)d_in[5];
    const float* b3 = (const float*)d_in[6];
    const float* W4 = (const float*)d_in[7];
    const float* b4 = (const float*)d_in[8];
    float* out = (float*)d_out;

    const int B  = 4096;
    const int D0 = 4096, D1 = 1024, D2 = 512, D3 = 256, D4 = N_WVL * 5; // 160

    // Workspace layout (bytes), all offsets well aligned.
    char* ws = (char*)d_ws;
    size_t o = 0;
    bf16_t* xb  = (bf16_t*)(ws + o); o += (size_t)B  * D0 * 2;  // 32 MiB
    bf16_t* w1b = (bf16_t*)(ws + o); o += (size_t)D1 * D0 * 2;  //  8 MiB
    bf16_t* w2b = (bf16_t*)(ws + o); o += (size_t)D2 * D1 * 2;  //  1 MiB
    bf16_t* w3b = (bf16_t*)(ws + o); o += (size_t)D3 * D2 * 2;  // 256 KiB
    bf16_t* w4b = (bf16_t*)(ws + o); o += (size_t)D4 * D3 * 2;  //  80 KiB
    bf16_t* h1b = (bf16_t*)(ws + o); o += (size_t)B  * D1 * 2;  //  8 MiB
    bf16_t* h2b = (bf16_t*)(ws + o); o += (size_t)B  * D2 * 2;  //  4 MiB
    bf16_t* h3b = (bf16_t*)(ws + o); o += (size_t)B  * D3 * 2;  //  2 MiB
    float*  prm = (float*) (ws + o); o += (size_t)B  * D4 * 4;  // 2.5 MiB

    // --- fp32 -> bf16 conversions ---
    auto cvt = [&](const float* src, bf16_t* dst, size_t n) {
        int n4 = (int)(n / 4);
        cvt_f32_bf16<<<(n4 + 255) / 256, 256, 0, stream>>>(src, dst, n4);
    };
    cvt(x,  xb,  (size_t)B  * D0);
    cvt(W1, w1b, (size_t)D1 * D0);
    cvt(W2, w2b, (size_t)D2 * D1);
    cvt(W3, w3b, (size_t)D3 * D2);
    cvt(W4, w4b, (size_t)D4 * D3);

    // --- Encoder MLP ---
    // L1: (B,4096)x(1024,4096)^T -> relu -> bf16   [LDS-tiled, async copies]
    gemm_bf16_wmma_lds<true><<<(B / MBLK) * (D1 / NBLK), 256, 0, stream>>>(
        xb, w1b, b1, h1b, B, D1, D0);
    // L2: (B,1024)x(512,1024)^T -> relu -> bf16
    gemm_bf16_wmma_lds<true><<<(B / MBLK) * (D2 / NBLK), 256, 0, stream>>>(
        h1b, w2b, b2, h2b, B, D2, D1);
    // L3: (B,512)x(256,512)^T -> relu -> bf16
    gemm_bf16_wmma_lds<true><<<(B / MBLK) * (D3 / NBLK), 256, 0, stream>>>(
        h2b, w3b, b3, h3b, B, D3, D2);
    // L4: (B,256)x(160,256)^T -> f32 params (32-wide tiles, direct kernel)
    gemm_bf16_wmma<2, 2, false, false><<<(B / 32) * (D4 / 32), 32, 0, stream>>>(
        h3b, w4b, b4, prm, B, D4, D3);

    // --- Gabor synthesis ---
    gabor_synth<<<B, 256, 0, stream>>>(prm, out);
}
